// Model_90546500534468
// MI455X (gfx1250) — matrix operand order repaired and checked by reading, compiled, not run on hardware
//
#include <hip/hip_runtime.h>

// ---------------- dims ----------------
#define Bn    8
#define Sn    512
#define ENCn  7
#define MARKn 4
#define Dn    512
#define In    1024
#define Kc    4
#define NHn   8
#define DHn   128
#define NBn   256
#define Ln    4
#define PREDn 96
#define COUTn 7

typedef __attribute__((ext_vector_type(16))) __bf16 v16bf;
typedef __attribute__((ext_vector_type(8)))  __bf16 v8bf;
typedef __attribute__((ext_vector_type(8)))  float  v8f;

// pointer types for the gfx1250 async global->LDS builtin
typedef int v4i_vs __attribute__((vector_size(16)));
#if defined(__AMDGCN__)
typedef __attribute__((address_space(1))) v4i_vs* async_gptr;
typedef __attribute__((address_space(3))) v4i_vs* async_lptr;
#endif

#if defined(__AMDGCN__) && __has_builtin(__builtin_amdgcn_global_load_async_to_lds_b128)
#define HAVE_ASYNC_LDS 1
#else
#define HAVE_ASYNC_LDS 0
#endif

__device__ __forceinline__ void wait_async_zero() {
#if defined(__AMDGCN__)
#if __has_builtin(__builtin_amdgcn_s_wait_asynccnt)
  __builtin_amdgcn_s_wait_asynccnt(0);
#else
  asm volatile("s_wait_asynccnt 0" ::: "memory");
#endif
#endif
}

// ---------------- WMMA fragment helpers (bf16, 16x16x32) ----------------
// A fragment (16x32, row-major source, row stride ld elements):
//   lanes 0-15: row=lane,   K = [0..7] and [16..23]
//   lanes16-31: row=lane-16,K = [8..15] and [24..31]
__device__ __forceinline__ v16bf load_frag_a(const __bf16* tile, int ld) {
  int lane = threadIdx.x & 31;
  int row  = lane & 15;
  int k0   = (lane & 16) ? 8 : 0;
  const __bf16* p = tile + row * ld + k0;
  v8bf lo = *(const v8bf*)(p);
  v8bf hi = *(const v8bf*)(p + 16);
  v16bf r;
#pragma unroll
  for (int i = 0; i < 8; ++i) { r[i] = lo[i]; r[i + 8] = hi[i]; }
  return r;
}

// B fragment (32x16). Source is W^T row-major [N][K] so each lane reads its
// column n = lane&15 contiguously along K:
//   lanes 0-15: K = [0..7] and [16..23]; lanes 16-31: K = [8..15] and [24..31]
__device__ __forceinline__ v16bf load_frag_b(const __bf16* tile, int ld) {
  int lane = threadIdx.x & 31;
  int row  = lane & 15;                  // n within 16-wide subtile
  int k0   = (lane & 16) ? 8 : 0;
  const __bf16* p = tile + row * ld + k0;
  v8bf lo = *(const v8bf*)(p);
  v8bf hi = *(const v8bf*)(p + 16);
  v16bf r;
#pragma unroll
  for (int i = 0; i < 8; ++i) { r[i] = lo[i]; r[i + 8] = hi[i]; }
  return r;
}

__device__ __forceinline__ v8f wmma_bf16(v16bf a, v16bf b, v8f c) {
  return __builtin_amdgcn_wmma_f32_16x16x32_bf16(false, a, false, b, (short)0, c,
                                                 false, false);
}

// ---------------- weight convert: fp32 [K,N] -> bf16 [N,K] ----------------
__global__ void wt_cvt(const float* __restrict__ W, __bf16* __restrict__ Wt,
                       int Kd, int N) {
  int idx = blockIdx.x * blockDim.x + threadIdx.x;
  if (idx >= Kd * N) return;
  int n = idx % N, k = idx / N;
  Wt[(size_t)n * Kd + k] = (__bf16)W[idx];
}

// ---------------- embedding ----------------
__global__ void embed_kernel(const float* __restrict__ xe, const float* __restrict__ xm,
                             const float* __restrict__ W, const float* __restrict__ bW,
                             float* __restrict__ x) {
  int idx = blockIdx.x * blockDim.x + threadIdx.x;
  if (idx >= Bn * Sn * Dn) return;
  int d = idx & (Dn - 1);
  int bs = idx >> 9;
  float acc = bW[d];
#pragma unroll
  for (int i = 0; i < ENCn; ++i) acc += xe[bs * ENCn + i] * W[i * Dn + d];
#pragma unroll
  for (int i = 0; i < MARKn; ++i) acc += xm[bs * MARKn + i] * W[(ENCn + i) * Dn + d];
  x[idx] = acc;
}

// ---------------- layernorm -> bf16 ----------------
__global__ __launch_bounds__(256)
void layernorm_to_bf16(const float* __restrict__ x, const float* __restrict__ w,
                       __bf16* __restrict__ out) {
  int row = blockIdx.x;
  int t = threadIdx.x;
  const float* xr = x + (size_t)row * Dn;
  float v0 = xr[t], v1 = xr[t + 256];
  float s = v0 + v1, ss = v0 * v0 + v1 * v1;
  __shared__ float as_[8], bs_[8];
  int lane = t & 31, wv = t >> 5;
#pragma unroll
  for (int off = 16; off; off >>= 1) {
    s += __shfl_xor(s, off, 32);
    ss += __shfl_xor(ss, off, 32);
  }
  if (lane == 0) { as_[wv] = s; bs_[wv] = ss; }
  __syncthreads();
  s = 0.f; ss = 0.f;
#pragma unroll
  for (int i = 0; i < 8; ++i) { s += as_[i]; ss += bs_[i]; }
  float mu = s * (1.f / Dn);
  float var = ss * (1.f / Dn) - mu * mu;
  float r = rsqrtf(var + 1e-5f);
  out[(size_t)row * Dn + t]       = (__bf16)((v0 - mu) * r * w[t]);
  out[(size_t)row * Dn + t + 256] = (__bf16)((v1 - mu) * r * w[t + 256]);
}

// ---------------- GEMM: C[M,N] = A[M,K](bf16) @ Wt[N,K](bf16) + bias (+Cin) ----
// A tile (16 rows x Kd) staged to LDS via gfx1250 async loads; k-loop manually
// 2x unrolled with ping-pong fragment registers (no rotation copies): each
// phase's WMMAs overlap the other phase's loads.
// Grids are exact: every block has full work (no early exits before barriers).
__global__ __launch_bounds__(256)
void gemm_bf16_wmma(const __bf16* __restrict__ A, const __bf16* __restrict__ Wt,
                    const float* __restrict__ bias, const float* __restrict__ Cin,
                    float* __restrict__ Cout, int M, int N, int Kd) {
  extern __shared__ __align__(16) __bf16 As[];   // [16][Kd]
  int wave = threadIdx.x >> 5;
  int lane = threadIdx.x & 31;
  int m0 = blockIdx.y * 16;
  int n0 = (blockIdx.x * 8 + wave) * 64;

  // ---- stage A tile into LDS (16*Kd bf16) ----
  const __bf16* Ab = A + (size_t)m0 * Kd;
  int chunks = (16 * Kd) / 8;                    // 8 bf16 (16B) per chunk
  for (int c = threadIdx.x; c < chunks; c += 256) {
    int off = c * 8;
#if HAVE_ASYNC_LDS
    __builtin_amdgcn_global_load_async_to_lds_b128(
        (async_gptr)(Ab + off), (async_lptr)(As + off), 0, 0);
#else
    *(v8bf*)(As + off) = *(const v8bf*)(Ab + off);
#endif
  }
#if HAVE_ASYNC_LDS
  wait_async_zero();
#endif
  __syncthreads();

  // ---- k-loop, 2x unrolled ping-pong fragments ----
  v8f acc[4] = {};
  const __bf16* Wb = Wt + (size_t)n0 * Kd;

  v16bf a0 = load_frag_a(As, Kd);
  v16bf b0[4];
#pragma unroll
  for (int t = 0; t < 4; ++t) b0[t] = load_frag_b(Wb + (size_t)(t * 16) * Kd, Kd);
  v16bf a1, b1[4];

  for (int k = 0; k < Kd; k += 64) {
    // issue phase-1 loads (k+32)
    __builtin_prefetch(Wb + k + 256, 0, 1);
    a1 = load_frag_a(As + k + 32, Kd);
#pragma unroll
    for (int t = 0; t < 4; ++t)
      b1[t] = load_frag_b(Wb + (size_t)(t * 16) * Kd + k + 32, Kd);
    // compute phase-0
#pragma unroll
    for (int t = 0; t < 4; ++t) acc[t] = wmma_bf16(a0, b0[t], acc[t]);
    // issue phase-0 loads for next iteration (k+64)
    if (k + 64 < Kd) {
      a0 = load_frag_a(As + k + 64, Kd);
#pragma unroll
      for (int t = 0; t < 4; ++t)
        b0[t] = load_frag_b(Wb + (size_t)(t * 16) * Kd + k + 64, Kd);
    }
    // compute phase-1
#pragma unroll
    for (int t = 0; t < 4; ++t) acc[t] = wmma_bf16(a1, b1[t], acc[t]);
  }

  int half8 = (lane & 16) ? 8 : 0;
  int nl = lane & 15;
#pragma unroll
  for (int t = 0; t < 4; ++t) {
    int n = n0 + t * 16 + nl;
    float bv = bias ? bias[n] : 0.f;
#pragma unroll
    for (int i = 0; i < 8; ++i) {
      int m = m0 + i + half8;
      float v = acc[t][i] + bv;
      if (Cin) v += Cin[(size_t)m * N + n];
      Cout[(size_t)m * N + n] = v;
    }
  }
}

// ---------------- causal depthwise conv (K=4) + SiLU ----------------
__global__ void conv_silu(const float* __restrict__ u, const float* __restrict__ cw,
                          const float* __restrict__ cb, float* __restrict__ xc) {
  int idx = blockIdx.x * blockDim.x + threadIdx.x;
  if (idx >= Bn * Sn * In) return;
  int c = idx & (In - 1);
  int bs = idx >> 10;
  int s = bs & (Sn - 1);
  float acc = cb[c];
#pragma unroll
  for (int k = 0; k < Kc; ++k) {
    int sp = s - (Kc - 1) + k;
    if (sp >= 0) acc += u[(size_t)(bs - s + sp) * (2 * In) + c] * cw[c * Kc + k];
  }
  xc[idx] = acc / (1.f + __expf(-acc)); // silu
}

// ---------------- headwise 4x4 block projections + pack bf16 ----------------
__global__ void headwise_pack(const float* __restrict__ u, const float* __restrict__ xc,
                              const float* __restrict__ Wq, const float* __restrict__ Wk,
                              const float* __restrict__ Wv,
                              __bf16* __restrict__ Qb, __bf16* __restrict__ Kb,
                              __bf16* __restrict__ Vt) {
  int idx = blockIdx.x * blockDim.x + threadIdx.x;
  if (idx >= Bn * Sn * In) return;
  int c = idx & (In - 1);
  int bs = idx >> 10;
  int s = bs & (Sn - 1);
  int b = bs >> 9;
  int nb = c >> 2, o = c & 3;
  const float* xcb = xc + (size_t)bs * In + nb * 4;
  const float* xmb = u + (size_t)bs * (2 * In) + nb * 4;
  const float* wq = Wq + (nb * 4 + o) * 4;
  const float* wk = Wk + (nb * 4 + o) * 4;
  const float* wv = Wv + (nb * 4 + o) * 4;
  float q = 0.f, k = 0.f, v = 0.f;
#pragma unroll
  for (int i = 0; i < 4; ++i) {
    q += wq[i] * xcb[i];
    k += wk[i] * xcb[i];
    v += wv[i] * xmb[i];
  }
  int hh = c >> 7, d = c & (DHn - 1);
  size_t bh = (size_t)(b * NHn + hh);
  Qb[(bh * Sn + s) * DHn + d] = (__bf16)q;
  Kb[(bh * Sn + s) * DHn + d] = (__bf16)k;
  Vt[(bh * DHn + d) * Sn + s] = (__bf16)v;
}

// ---------------- gate projections ig/fg: one wave per (b,h,s) ----------------
__global__ __launch_bounds__(256)
void gates_kernel(const __bf16* __restrict__ Qb, const __bf16* __restrict__ Kb,
                  const __bf16* __restrict__ Vt,
                  const float* __restrict__ igW, const float* __restrict__ igb,
                  const float* __restrict__ fgW, const float* __restrict__ fgb,
                  float* __restrict__ ig, float* __restrict__ fg) {
  int wid = blockIdx.x * 8 + (threadIdx.x >> 5);
  int lane = threadIdx.x & 31;
  int s = wid & (Sn - 1);
  int h = (wid >> 9) & (NHn - 1);
  int b = wid >> 12;
  float ai = 0.f, af = 0.f;
  for (int c = lane; c < In; c += 32) {
    int ch = c >> 7, cd = c & (DHn - 1);
    size_t bhc = (size_t)(b * NHn + ch);
    float qv = (float)Qb[(bhc * Sn + s) * DHn + cd];
    float kv = (float)Kb[(bhc * Sn + s) * DHn + cd];
    float vv = (float)Vt[(bhc * DHn + cd) * Sn + s];
    ai += qv * igW[c * NHn + h] + kv * igW[(In + c) * NHn + h] +
          vv * igW[(2 * In + c) * NHn + h];
    af += qv * fgW[c * NHn + h] + kv * fgW[(In + c) * NHn + h] +
          vv * fgW[(2 * In + c) * NHn + h];
  }
#pragma unroll
  for (int off = 16; off; off >>= 1) {
    ai += __shfl_xor(ai, off, 32);
    af += __shfl_xor(af, off, 32);
  }
  if (lane == 0) {
    size_t o = (size_t)(b * NHn + h) * Sn + s;
    ig[o] = ai + igb[h];
    fg[o] = af + fgb[h];
  }
}

// ---------------- log-sigmoid cumsum per (b,h) ----------------
__global__ __launch_bounds__(512)
void lfc_scan(const float* __restrict__ fg, float* __restrict__ lfc) {
  int bh = blockIdx.x;
  int t = threadIdx.x;
  __shared__ float sbuf[Sn];
  float f = fg[(size_t)bh * Sn + t];
  float lsig = (f >= 0.f) ? -log1pf(__expf(-f)) : (f - log1pf(__expf(f)));
  sbuf[t] = lsig;
  __syncthreads();
  for (int off = 1; off < Sn; off <<= 1) {
    float add = (t >= off) ? sbuf[t - off] : 0.f;
    __syncthreads();
    sbuf[t] += add;
    __syncthreads();
  }
  if (t == 0) lfc[(size_t)bh * (Sn + 1)] = 0.f;
  lfc[(size_t)bh * (Sn + 1) + t + 1] = sbuf[t];
}

// ---------------- mLSTM stabilized attention + per-head norm ----------------
// one wave per (b,h, 16-row query block); flash-style over 32-wide key blocks
__global__ __launch_bounds__(32)
void mlstm_attn(const __bf16* __restrict__ Qb, const __bf16* __restrict__ Kb,
                const __bf16* __restrict__ Vt, const float* __restrict__ lfc,
                const float* __restrict__ ig, const float* __restrict__ onw,
                float* __restrict__ hn) {
  int bh = blockIdx.y;
  int qb = blockIdx.x;
  int lane = threadIdx.x;
  int half = lane >> 4;
  int nl = lane & 15;
  int q0 = qb * 16;
  int hh = bh & (NHn - 1);
  int b = bh >> 3;

  const __bf16* Qh = Qb + (size_t)bh * Sn * DHn;
  const __bf16* Kh = Kb + (size_t)bh * Sn * DHn;
  const __bf16* Vh = Vt + (size_t)bh * DHn * Sn;
  const float* lf = lfc + (size_t)bh * (Sn + 1);
  const float* igp = ig + (size_t)bh * Sn;

  v16bf qf[4];
#pragma unroll
  for (int kk = 0; kk < 4; ++kk)
    qf[kk] = load_frag_a(Qh + (size_t)q0 * DHn + kk * 32, DHn);

  float lfr[8];
#pragma unroll
  for (int i = 0; i < 8; ++i) lfr[i] = lf[q0 + i + 8 * half + 1];

  v8f oacc[8] = {};
  float mrun[8], csum[8];
#pragma unroll
  for (int i = 0; i < 8; ++i) { mrun[i] = -1e30f; csum[i] = 0.f; }

  __shared__ __align__(16) __bf16 cs[16 * 32];

  const float scale = 0.088388347648318447f; // DH^-0.5
  int kbmax = (q0 + 15) / 32;
  for (int kb = 0; kb <= kbmax; ++kb) {
    int t0c = kb * 32;
    v8f s0 = {}, s1 = {};
#pragma unroll
    for (int kk = 0; kk < 4; ++kk) {
      v16bf b0 = load_frag_b(Kh + (size_t)t0c * DHn + kk * 32, DHn);
      v16bf b1 = load_frag_b(Kh + (size_t)(t0c + 16) * DHn + kk * 32, DHn);
      s0 = wmma_bf16(qf[kk], b0, s0);
      s1 = wmma_bf16(qf[kk], b1, s1);
    }
    int t0 = t0c + nl, t1 = t0 + 16;
    float l0 = lf[t0 + 1], g0 = igp[t0];
    float l1 = lf[t1 + 1], g1 = igp[t1];
#pragma unroll
    for (int i = 0; i < 8; ++i) {
      int srow = q0 + i + 8 * half;
      float ld0 = (t0 <= srow) ? (lfr[i] - l0 + g0) : -1e30f;
      float ld1 = (t1 <= srow) ? (lfr[i] - l1 + g1) : -1e30f;
      float bm = fmaxf(ld0, ld1);
#pragma unroll
      for (int off = 1; off < 16; off <<= 1) bm = fmaxf(bm, __shfl_xor(bm, off, 16));
      float newm = fmaxf(mrun[i], bm);
      float fac = __expf(mrun[i] - newm);
      mrun[i] = newm;
      float c0 = s0[i] * scale * __expf(ld0 - newm);
      float c1 = s1[i] * scale * __expf(ld1 - newm);
      float bs = c0 + c1;
#pragma unroll
      for (int off = 1; off < 16; off <<= 1) bs += __shfl_xor(bs, off, 16);
      csum[i] = csum[i] * fac + bs;
#pragma unroll
      for (int j = 0; j < 8; ++j) oacc[j][i] *= fac;
      cs[(i + 8 * half) * 32 + nl] = (__bf16)c0;
      cs[(i + 8 * half) * 32 + 16 + nl] = (__bf16)c1;
    }
    __syncthreads();
    v16bf af = load_frag_a(cs, 32);
#pragma unroll
    for (int j = 0; j < 8; ++j) {
      v16bf bv = load_frag_b(Vh + (size_t)(j * 16) * Sn + t0c, Sn);
      oacc[j] = wmma_bf16(af, bv, oacc[j]);
    }
    __syncthreads();
  }

  // normalize by n = max(|sum C|, exp(-maxD)), then per-head layernorm
#pragma unroll
  for (int i = 0; i < 8; ++i) {
    float nrm = fmaxf(fabsf(csum[i]), __expf(-mrun[i]));
    float inv = 1.f / (nrm + 1e-6f);
    float s1v = 0.f, s2v = 0.f;
#pragma unroll
    for (int j = 0; j < 8; ++j) {
      float v = oacc[j][i] * inv;
      oacc[j][i] = v;
      s1v += v;
      s2v += v * v;
    }
#pragma unroll
    for (int off = 1; off < 16; off <<= 1) {
      s1v += __shfl_xor(s1v, off, 16);
      s2v += __shfl_xor(s2v, off, 16);
    }
    float mu = s1v * (1.f / DHn);
    float var = s2v * (1.f / DHn) - mu * mu;
    float r = rsqrtf(var + 1e-5f);
    int srow = q0 + i + 8 * half;
    float* dst = hn + ((size_t)(b * Sn + srow)) * In + hh * DHn;
#pragma unroll
    for (int j = 0; j < 8; ++j) {
      int d = j * 16 + nl;
      dst[d] = (oacc[j][i] - mu) * r * onw[hh * DHn + d];
    }
  }
}

// ---------------- skip + output gate, emit bf16 for down GEMM ----------------
__global__ void fuse_gate(const float* __restrict__ hn, const float* __restrict__ xc,
                          const float* __restrict__ u, const float* __restrict__ skw,
                          __bf16* __restrict__ hhb) {
  int idx = blockIdx.x * blockDim.x + threadIdx.x;
  if (idx >= Bn * Sn * In) return;
  int c = idx & (In - 1);
  int bs = idx >> 10;
  float z = u[(size_t)bs * (2 * In) + In + c];
  float sz = z / (1.f + __expf(-z));
  hhb[idx] = (__bf16)((hn[idx] + skw[c] * xc[idx]) * sz);
}

// ---------------- post-norm + prediction head ----------------
__global__ __launch_bounds__(256)
void post_head(const float* __restrict__ x, const float* __restrict__ pw,
               const float* __restrict__ hW, const float* __restrict__ hb,
               float* __restrict__ out) {
  int bp = blockIdx.x;
  int b = bp / PREDn, p = bp % PREDn;
  int s = Sn - PREDn + p;
  const float* xr = x + ((size_t)b * Sn + s) * Dn;
  int t = threadIdx.x;
  float v0 = xr[t], v1 = xr[t + 256];
  float sm = v0 + v1, ssq = v0 * v0 + v1 * v1;
  __shared__ float as_[8], bs_[8];
  __shared__ float xn[Dn];
  int lane = t & 31, wv = t >> 5;
#pragma unroll
  for (int off = 16; off; off >>= 1) {
    sm += __shfl_xor(sm, off, 32);
    ssq += __shfl_xor(ssq, off, 32);
  }
  if (lane == 0) { as_[wv] = sm; bs_[wv] = ssq; }
  __syncthreads();
  sm = 0.f; ssq = 0.f;
#pragma unroll
  for (int i = 0; i < 8; ++i) { sm += as_[i]; ssq += bs_[i]; }
  float mu = sm * (1.f / Dn);
  float var = ssq * (1.f / Dn) - mu * mu;
  float r = rsqrtf(var + 1e-5f);
  xn[t] = (v0 - mu) * r * pw[t];
  xn[t + 256] = (v1 - mu) * r * pw[t + 256];
  __syncthreads();
  if (t < COUTn) {
    float a = hb[t];
    for (int d = 0; d < Dn; ++d) a += xn[d] * hW[d * COUTn + t];
    out[(size_t)bp * COUTn + t] = a;
  }
}

// ---------------- host launcher ----------------
extern "C" void kernel_launch(void* const* d_in, const int* in_sizes, int n_in,
                              void* d_out, int out_size, void* d_ws, size_t ws_size,
                              hipStream_t stream) {
  const float* x_enc   = (const float*)d_in[0];
  const float* x_mark  = (const float*)d_in[1];
  const float* emb_W   = (const float*)d_in[4];
  const float* emb_b   = (const float*)d_in[5];
  const float* blk_ln  = (const float*)d_in[6];
  const float* up_W    = (const float*)d_in[7];
  const float* up_b    = (const float*)d_in[8];
  const float* conv_w  = (const float*)d_in[9];
  const float* conv_b  = (const float*)d_in[10];
  const float* Wq      = (const float*)d_in[11];
  const float* Wk      = (const float*)d_in[12];
  const float* Wv      = (const float*)d_in[13];
  const float* ig_W    = (const float*)d_in[14];
  const float* ig_b    = (const float*)d_in[15];
  const float* fg_W    = (const float*)d_in[16];
  const float* fg_b    = (const float*)d_in[17];
  const float* onorm_w = (const float*)d_in[18];
  const float* skip_w  = (const float*)d_in[19];
  const float* down_W  = (const float*)d_in[20];
  const float* down_b  = (const float*)d_in[21];
  const float* post_w  = (const float*)d_in[22];
  const float* head_W  = (const float*)d_in[23];
  const float* head_b  = (const float*)d_in[24];

  char* p = (char*)d_ws;
  auto alloc = [&](size_t bytes) -> char* {
    char* r = p;
    p += (bytes + 255) & ~(size_t)255;
    return r;
  };
  __bf16* upWT = (__bf16*)alloc((size_t)Ln * 2 * In * Dn * 2);
  __bf16* dWT  = (__bf16*)alloc((size_t)Ln * Dn * In * 2);
  float*  x    = (float*)alloc((size_t)Bn * Sn * Dn * 4);
  __bf16* hbf  = (__bf16*)alloc((size_t)Bn * Sn * Dn * 2);
  float*  u    = (float*)alloc((size_t)Bn * Sn * 2 * In * 4);
  float*  xc   = (float*)alloc((size_t)Bn * Sn * In * 4);
  __bf16* Qb   = (__bf16*)alloc((size_t)Bn * NHn * Sn * DHn * 2);
  __bf16* Kb   = (__bf16*)alloc((size_t)Bn * NHn * Sn * DHn * 2);
  __bf16* Vt   = (__bf16*)alloc((size_t)Bn * NHn * DHn * Sn * 2);
  float*  ig   = (float*)alloc((size_t)Bn * NHn * Sn * 4);
  float*  fg   = (float*)alloc((size_t)Bn * NHn * Sn * 4);
  float*  lfc  = (float*)alloc((size_t)Bn * NHn * (Sn + 1) * 4);
  float*  hn   = (float*)alloc((size_t)Bn * Sn * In * 4);
  __bf16* hhb  = (__bf16*)alloc((size_t)Bn * Sn * In * 2);

  // convert + transpose all GEMM weights to bf16 [N][K] up front
  for (int l = 0; l < Ln; ++l) {
    wt_cvt<<<(Dn * 2 * In + 255) / 256, 256, 0, stream>>>(
        up_W + (size_t)l * Dn * 2 * In, upWT + (size_t)l * 2 * In * Dn, Dn, 2 * In);
    wt_cvt<<<(In * Dn + 255) / 256, 256, 0, stream>>>(
        down_W + (size_t)l * In * Dn, dWT + (size_t)l * Dn * In, In, Dn);
  }

  embed_kernel<<<(Bn * Sn * Dn + 255) / 256, 256, 0, stream>>>(x_enc, x_mark, emb_W,
                                                               emb_b, x);

  for (int l = 0; l < Ln; ++l) {
    layernorm_to_bf16<<<Bn * Sn, 256, 0, stream>>>(x, blk_ln + l * Dn, hbf);

    dim3 gUp(2 * In / 512, Bn * Sn / 16);
    gemm_bf16_wmma<<<gUp, 256, 16 * Dn * 2, stream>>>(
        hbf, upWT + (size_t)l * 2 * In * Dn, up_b + l * 2 * In, nullptr, u,
        Bn * Sn, 2 * In, Dn);

    conv_silu<<<(Bn * Sn * In + 255) / 256, 256, 0, stream>>>(
        u, conv_w + l * In * Kc, conv_b + l * In, xc);

    headwise_pack<<<(Bn * Sn * In + 255) / 256, 256, 0, stream>>>(
        u, xc, Wq + (size_t)l * NBn * 16, Wk + (size_t)l * NBn * 16,
        Wv + (size_t)l * NBn * 16, Qb, Kb, Vt);

    gates_kernel<<<Bn * NHn * Sn / 8, 256, 0, stream>>>(
        Qb, Kb, Vt, ig_W + (size_t)l * 3 * In * NHn, ig_b + l * NHn,
        fg_W + (size_t)l * 3 * In * NHn, fg_b + l * NHn, ig, fg);

    lfc_scan<<<Bn * NHn, 512, 0, stream>>>(fg, lfc);

    dim3 gAtt(Sn / 16, Bn * NHn);
    mlstm_attn<<<gAtt, 32, 0, stream>>>(Qb, Kb, Vt, lfc, ig, onorm_w + l * In, hn);

    fuse_gate<<<(Bn * Sn * In + 255) / 256, 256, 0, stream>>>(hn, xc, u,
                                                              skip_w + l * In, hhb);

    dim3 gDn(Dn / 512, Bn * Sn / 16);
    gemm_bf16_wmma<<<gDn, 256, 16 * In * 2, stream>>>(
        hhb, dWT + (size_t)l * Dn * In, down_b + l * Dn, x, x, Bn * Sn, Dn, In);
  }

  post_head<<<Bn * PREDn, 256, 0, stream>>>(x, post_w, head_W, head_b, (float*)d_out);
}
